// SimpleAttention_14027363188928
// MI455X (gfx1250) — matrix operand order repaired and checked by reading, not compiled, not run on hardware
//
#include <hip/hip_runtime.h>

typedef __attribute__((ext_vector_type(16))) _Float16 v16h;
typedef __attribute__((ext_vector_type(8)))  _Float16 v8h;
typedef __attribute__((ext_vector_type(4)))  _Float16 v4h;
typedef __attribute__((ext_vector_type(2)))  _Float16 v2h;
typedef __attribute__((ext_vector_type(8)))  float    v8f;
typedef __attribute__((ext_vector_type(4)))  unsigned int u32x4;
typedef __attribute__((ext_vector_type(8)))  int      i32x8;
typedef __attribute__((ext_vector_type(4)))  int      i32x4;

#define BATCH 16
#define SEQ   2048
#define DIM   256
#define KT    64      // keys per block iteration
#define WPB   4       // waves per block (each wave owns 16 query rows)
#define TPB   128

// LDS carve-up (halfs): kt[KT][DIM], vt[DIM][KT], qs[WPB][16][DIM], pst[WPB][16][KT]
#define KT_OFF   0
#define VT_OFF   (KT * DIM)
#define QS_OFF   (VT_OFF + DIM * KT)
#define PST_OFF  (QS_OFF + WPB * 16 * DIM)
#define SMEM_HALFS (PST_OFF + WPB * 16 * KT)

// packed f32x2 -> f16x2 (v_cvt_pk_rtz_f16_f32), bit-cast to _Float16 vector
static __device__ __forceinline__ v2h pk2(float a, float b) {
  return __builtin_bit_cast(v2h, __builtin_amdgcn_cvt_pkrtz(a, b));
}

// ---------------- pre-pass 1: 1/max(||row||, eps) ----------------
__global__ __launch_bounds__(256) void rnorm_kernel(const float* __restrict__ x,
                                                    float* __restrict__ rn) {
  const int rows = BATCH * SEQ;
  int gw   = (int)((blockIdx.x * blockDim.x + threadIdx.x) >> 5);
  int lane = threadIdx.x & 31;
  if (gw >= rows) return;
  const float* row = x + (size_t)gw * DIM;
  float s = 0.f;
#pragma unroll
  for (int i = 0; i < DIM / 32; ++i) {
    float v = row[lane + i * 32];
    s += v * v;
  }
#pragma unroll
  for (int off = 16; off > 0; off >>= 1) s += __shfl_xor(s, off, 32);
  if (lane == 0) rn[gw] = 1.0f / fmaxf(sqrtf(s), 1e-12f);
}

// ---------------- pre-pass 2: kh = f16(normalize(x)), vT = f16(x)^T ----------------
#define CT 64   // n-rows per conv block
__global__ __launch_bounds__(256) void conv_kernel(const float* __restrict__ x,
                                                   const float* __restrict__ rn,
                                                   _Float16* __restrict__ kh,
                                                   _Float16* __restrict__ vT) {
  __shared__ _Float16 tile[CT][DIM];   // raw f16 tile for the transpose (32 KB)
  const int b  = blockIdx.x >> 5;      // SEQ/CT = 32 blocks per batch
  const int n0 = (blockIdx.x & 31) * CT;
  const int t  = threadIdx.x;
  {
    const int r  = t >> 2;             // row in tile 0..63
    const int d0 = (t & 3) * 64;       // 64 d-elements per thread
    const float* src = x + ((size_t)b * SEQ + n0 + r) * DIM + d0;
    const float  rv  = rn[(size_t)b * SEQ + n0 + r];
    _Float16* khrow  = kh + ((size_t)b * SEQ + n0 + r) * DIM + d0;
#pragma unroll
    for (int i = 0; i < 64; i += 4) {
      float4 v = *(const float4*)(src + i);
      v2h a01 = pk2(v.x * rv, v.y * rv);
      v2h a23 = pk2(v.z * rv, v.w * rv);
      *(v4h*)(khrow + i) = (v4h){a01.x, a01.y, a23.x, a23.y};
      v2h r01 = pk2(v.x, v.y);
      v2h r23 = pk2(v.z, v.w);
      *(v4h*)&tile[r][d0 + i] = (v4h){r01.x, r01.y, r23.x, r23.y};
    }
  }
  __syncthreads();
  // transposed write: thread t owns d-row t, writes vT[b][t][n0..n0+63]
  {
    _Float16* vrow = vT + ((size_t)b * DIM + t) * SEQ + n0;
#pragma unroll
    for (int j = 0; j < CT; j += 2) {
      v2h p; p.x = tile[j][t]; p.y = tile[j + 1][t];
      *(v2h*)(vrow + j) = p;
    }
  }
}

// ---------------- TDM 2D tile load: global(f16) -> LDS ----------------
#if __has_builtin(__builtin_amdgcn_tensor_load_to_lds)
#define HAVE_TDM 1
static __device__ __forceinline__ void tdm_load_2d(unsigned int lds_byte_off,
                                                   const void* gaddr,
                                                   unsigned int tensor_d0,
                                                   unsigned int tensor_d1,
                                                   unsigned int tile_d0,
                                                   unsigned int tile_d1,
                                                   unsigned long long stride0) {
  unsigned long long ga = (unsigned long long)(uintptr_t)gaddr;
  u32x4 g0;
  g0[0] = 1u;                                             // count=1 (valid user D#)
  g0[1] = lds_byte_off;                                   // lds_addr
  g0[2] = (unsigned int)(ga & 0xFFFFFFFFu);               // global_addr[31:0]
  g0[3] = (unsigned int)((ga >> 32) & 0x1FFFFFFu) | (2u << 30);  // addr[56:32], type=2
  i32x8 g1;
  g1[0] = (int)(1u << 16);                                // data_size=1 -> 2 bytes
  g1[1] = (int)((tensor_d0 & 0xFFFFu) << 16);             // tensor_dim0[15:0]
  g1[2] = (int)((tensor_d0 >> 16) | ((tensor_d1 & 0xFFFFu) << 16));
  g1[3] = (int)((tensor_d1 >> 16) | ((tile_d0 & 0xFFFFu) << 16));
  g1[4] = (int)(tile_d1 & 0xFFFFu);                       // tile_dim1, tile_dim2=0
  g1[5] = (int)(unsigned int)(stride0 & 0xFFFFFFFFu);     // tensor_dim0_stride[31:0]
  g1[6] = (int)(unsigned int)((stride0 >> 32) & 0xFFFFu); // stride0[47:32], stride1=0
  g1[7] = 0;
  i32x4 z4 = {0, 0, 0, 0};
  i32x8 z8 = {0, 0, 0, 0, 0, 0, 0, 0};
  __builtin_amdgcn_tensor_load_to_lds(g0, g1, z4, z4, z8, 0);
}
#else
#define HAVE_TDM 0
#endif

// ---------------- fused flash-attention ----------------
__global__ __launch_bounds__(TPB) void attn_kernel(const float* __restrict__ x,
                                                   const _Float16* __restrict__ kh,
                                                   const _Float16* __restrict__ vT,
                                                   float* __restrict__ out) {
  extern __shared__ _Float16 smem[];
  _Float16* ktb = smem + KT_OFF;    // normalized K tile, [key][DIM]
  _Float16* vtb = smem + VT_OFF;    // V tile transposed,  [d][KT]
  _Float16* qsb = smem + QS_OFF;    // per-wave Q tiles,   [wave][16][DIM]
  _Float16* psb = smem + PST_OFF;   // per-wave P staging, [wave][16][KT]

  const int tid  = threadIdx.x;
  const int wave = tid >> 5;
  const int lane = tid & 31;
  const int half = lane >> 4;       // 0: lanes 0-15, 1: lanes 16-31
  const int l16  = lane & 15;
  const int kb   = half * 16;       // K-half offset for B fragments
  const int base0 = half * 16;      // K-half offset for A fragments

  const int b  = blockIdx.x >> 5;                       // 32 blocks per batch
  const int q0 = ((blockIdx.x & 31) * WPB + wave) * 16; // this wave's query rows
  const float* xb = x + (size_t)b * SEQ * DIM;

  // ---- stage this wave's Q tile into LDS (f16, * d^-0.5), wave-private ----
  _Float16* qw = qsb + wave * (16 * DIM);
  {
    const float* qrow = xb + (size_t)(q0 + l16) * DIM + half * 128;
    _Float16* dst = qw + l16 * DIM + half * 128;
#pragma unroll
    for (int i = 0; i < 128; i += 4) {
      float4 v = *(const float4*)(qrow + i);
      v2h h01 = pk2(v.x * 0.0625f, v.y * 0.0625f);
      v2h h23 = pk2(v.z * 0.0625f, v.w * 0.0625f);
      *(v4h*)(dst + i) = (v4h){h01.x, h01.y, h23.x, h23.y};
    }
  }

  v8f acc[16];
#pragma unroll
  for (int ch = 0; ch < 16; ++ch) acc[ch] = (v8f){};
  float mrow[8], lrow[8];
#pragma unroll
  for (int r = 0; r < 8; ++r) { mrow[r] = -1e30f; lrow[r] = 0.0f; }

  _Float16* pw = psb + wave * (16 * KT);

  for (int k0 = 0; k0 < SEQ; k0 += KT) {
    __syncthreads();   // previous tile fully consumed
#if HAVE_TDM
    if (wave == 0) {
      // K tile: 64 rows x 256 halfs, rows contiguous in kh
      tdm_load_2d((unsigned int)(KT_OFF * 2), kh + ((size_t)b * SEQ + k0) * DIM,
                  DIM, SEQ, DIM, KT, DIM);
      // V^T tile: 256 rows x 64 halfs, row stride SEQ in vT
      tdm_load_2d((unsigned int)(VT_OFF * 2), vT + (size_t)b * DIM * SEQ + k0,
                  SEQ, DIM, KT, DIM, SEQ);
      __builtin_amdgcn_s_wait_tensorcnt(0);
    }
#else
    {
      // fallback: plain f16 b128 copies
      const _Float16* ksrc = kh + ((size_t)b * SEQ + k0) * DIM;  // 16K halfs contiguous
      const uint4* ks = (const uint4*)(ksrc) + tid * 2;
      uint4* kd = (uint4*)(ktb) + tid * 2;
      kd[0] = ks[0]; kd[1] = ks[1];
      const _Float16* vsrc = vT + (size_t)b * DIM * SEQ + k0;
#pragma unroll
      for (int j = 0; j < 2; ++j) {
        const int dr = tid * 2 + j;
        const uint4* vs = (const uint4*)(vsrc + (size_t)dr * SEQ);
        uint4* vd = (uint4*)(vtb + dr * KT);
        vd[0] = vs[0]; vd[1] = vs[1];
      }
    }
#endif
    __syncthreads();

    // ---- S = (Q/sqrt(d)) * K^T : four 16x16 tiles over 64 keys ----
    v8f s0 = (v8f){}, s1 = (v8f){}, s2 = (v8f){}, s3 = (v8f){};
#pragma unroll
    for (int c = 0; c < 8; ++c) {
      v8h qlo = *(const v8h*)(qw + l16 * DIM + c * 32 + base0);
      v8h qhi = *(const v8h*)(qw + l16 * DIM + c * 32 + 8 + base0);
      v16h qa;
#pragma unroll
      for (int h = 0; h < 8; ++h) { qa[h] = qlo[h]; qa[8 + h] = qhi[h]; }
      v16h b0 = *(const v16h*)(ktb + (l16     ) * DIM + c * 32 + kb);
      v16h b1 = *(const v16h*)(ktb + (l16 + 16) * DIM + c * 32 + kb);
      v16h b2 = *(const v16h*)(ktb + (l16 + 32) * DIM + c * 32 + kb);
      v16h b3 = *(const v16h*)(ktb + (l16 + 48) * DIM + c * 32 + kb);
      s0 = __builtin_amdgcn_wmma_f32_16x16x32_f16(false, qa, false, b0, (short)0, s0, false, false);
      s1 = __builtin_amdgcn_wmma_f32_16x16x32_f16(false, qa, false, b1, (short)0, s1, false, false);
      s2 = __builtin_amdgcn_wmma_f32_16x16x32_f16(false, qa, false, b2, (short)0, s2, false, false);
      s3 = __builtin_amdgcn_wmma_f32_16x16x32_f16(false, qa, false, b3, (short)0, s3, false, false);
    }

    // ---- online softmax update (row = r + 8*half in C layout) ----
    float alpha[8];
#pragma unroll
    for (int r = 0; r < 8; ++r) {
      float mx = fmaxf(fmaxf(s0[r], s1[r]), fmaxf(s2[r], s3[r]));
#pragma unroll
      for (int off = 8; off > 0; off >>= 1) mx = fmaxf(mx, __shfl_xor(mx, off, 32));
      float mn = fmaxf(mrow[r], mx);
      float a  = __expf(mrow[r] - mn);
      float p0 = __expf(s0[r] - mn);
      float p1 = __expf(s1[r] - mn);
      float p2 = __expf(s2[r] - mn);
      float p3 = __expf(s3[r] - mn);
      s0[r] = p0; s1[r] = p1; s2[r] = p2; s3[r] = p3;
      float rs = (p0 + p1) + (p2 + p3);
#pragma unroll
      for (int off = 8; off > 0; off >>= 1) rs += __shfl_xor(rs, off, 32);
      lrow[r]  = lrow[r] * a + rs;
      mrow[r]  = mn;
      alpha[r] = a;
    }
#pragma unroll
    for (int ch = 0; ch < 16; ++ch) {
#pragma unroll
      for (int r = 0; r < 8; ++r) acc[ch][r] *= alpha[r];
    }

    // ---- P: C-layout -> A-layout via per-wave LDS round-trip ----
#pragma unroll
    for (int r = 0; r < 8; ++r) {
      _Float16* pr = pw + (r + half * 8) * KT;
      pr[l16]      = (_Float16)s0[r];
      pr[16 + l16] = (_Float16)s1[r];
      pr[32 + l16] = (_Float16)s2[r];
      pr[48 + l16] = (_Float16)s3[r];
    }
    v16h pa0, pa1;
    {
      v8h a0 = *(const v8h*)(pw + l16 * KT + base0);
      v8h a1 = *(const v8h*)(pw + l16 * KT + 8 + base0);
      v8h a2 = *(const v8h*)(pw + l16 * KT + 32 + base0);
      v8h a3 = *(const v8h*)(pw + l16 * KT + 40 + base0);
#pragma unroll
      for (int h = 0; h < 8; ++h) {
        pa0[h] = a0[h]; pa0[8 + h] = a1[h];
        pa1[h] = a2[h]; pa1[8 + h] = a3[h];
      }
    }

    // ---- acc += P * V over all 16 d-chunks ----
#pragma unroll
    for (int ch = 0; ch < 16; ++ch) {
      v16h vb0 = *(const v16h*)(vtb + (ch * 16 + l16) * KT + kb);
      v16h vb1 = *(const v16h*)(vtb + (ch * 16 + l16) * KT + 32 + kb);
      acc[ch] = __builtin_amdgcn_wmma_f32_16x16x32_f16(false, pa0, false, vb0, (short)0, acc[ch], false, false);
      acc[ch] = __builtin_amdgcn_wmma_f32_16x16x32_f16(false, pa1, false, vb1, (short)0, acc[ch], false, false);
    }
  }

  // ---- epilogue: divide by softmax denominator, store fp32 ----
  float* ob = out + ((size_t)b * SEQ + q0) * DIM;
#pragma unroll
  for (int r = 0; r < 8; ++r) {
    const int   row = r + half * 8;
    const float inv = 1.0f / lrow[r];
#pragma unroll
    for (int ch = 0; ch < 16; ++ch) {
      ob[(size_t)row * DIM + ch * 16 + l16] = acc[ch][r] * inv;
    }
  }
}

extern "C" void kernel_launch(void* const* d_in, const int* in_sizes, int n_in,
                              void* d_out, int out_size, void* d_ws, size_t ws_size,
                              hipStream_t stream) {
  (void)in_sizes; (void)n_in; (void)out_size; (void)ws_size;
  const float* x   = (const float*)d_in[0];
  float*       out = (float*)d_out;

  // workspace layout: rn (128 KB) | kh (16.78 MB) | vT (16.78 MB)
  float*    rn = (float*)d_ws;
  _Float16* kh = (_Float16*)((char*)d_ws + 131072);
  _Float16* vT = (_Float16*)((char*)d_ws + 131072 + (size_t)BATCH * SEQ * DIM * 2);

  const int rows = BATCH * SEQ;
  rnorm_kernel<<<(rows * 32 + 255) / 256, 256, 0, stream>>>(x, rn);
  conv_kernel<<<BATCH * (SEQ / CT), 256, 0, stream>>>(x, rn, kh, vT);

  const size_t smem_bytes = (size_t)SMEM_HALFS * sizeof(_Float16);  // 104 KB
  (void)hipFuncSetAttribute(reinterpret_cast<const void*>(attn_kernel),
                            hipFuncAttributeMaxDynamicSharedMemorySize, (int)smem_bytes);
  attn_kernel<<<BATCH * (SEQ / (WPB * 16)), TPB, smem_bytes, stream>>>(x, kh, vT, out);
}
